// SPGGNNConv_59854664237659
// MI455X (gfx1250) — matrix operand order repaired and checked by reading, not compiled, not run on hardware
//
#include <hip/hip_runtime.h>
#include <hip/hip_bf16.h>

typedef __bf16 bf16_t;
typedef __attribute__((ext_vector_type(16))) __bf16 v16bf;
typedef __attribute__((ext_vector_type(8)))  float  v8f;

#define N_NODES_C 50000
#define N_EDGES_C 800000
#define HIDDEN_C  128
#define KTOT_C    256
#define BUCKETS_C 20

// ---------------------------------------------------------------------------
// Kernel 0: collapse dist_emb_table @ W2[128:256] into 20 scalars
// ---------------------------------------------------------------------------
__global__ void d2_kernel(const float* __restrict__ table,
                          const float* __restrict__ W2,
                          float* __restrict__ d2) {
    int b = threadIdx.x;
    if (b < BUCKETS_C) {
        float s = 0.f;
        for (int k = 0; k < HIDDEN_C; ++k)
            s += table[b * HIDDEN_C + k] * W2[HIDDEN_C + k];
        d2[b] = s;
    }
}

// ---------------------------------------------------------------------------
// Kernel 1: zero the accumulators (d_out reused as agg, att_agg in ws)
// ---------------------------------------------------------------------------
__global__ void zero_kernel(float* __restrict__ out, int n_out,
                            float* __restrict__ att_agg, int n_agg) {
    int i = blockIdx.x * blockDim.x + threadIdx.x;
    int stride = gridDim.x * blockDim.x;
    for (int j = i; j < n_out; j += stride) out[j] = 0.f;
    for (int j = i; j < n_agg; j += stride) att_agg[j] = 0.f;
}

// ---------------------------------------------------------------------------
// Kernel 2: fused edge GEMM (bf16 WMMA) + attention + weighted scatter
//   block = 256 threads = 8 waves; each wave owns 32 edges (two 16-edge
//   M-tiles) -> each B fragment feeds TWO WMMAs, halving LDS traffic per
//   FLOP and hiding ds latency behind paired WMMA issue.
//   LDS: W1^T as bf16  (128 rows(N) x 256 cols(K)) = 64 KB
// ---------------------------------------------------------------------------
__global__ __launch_bounds__(256)
void edge_kernel(const float* __restrict__ x,      // [N,128]
                 const int*   __restrict__ ei,     // [2,E]
                 const int*   __restrict__ dist,   // [E]
                 const float* __restrict__ W1,     // [256,128]
                 const float* __restrict__ W2,     // [256]
                 const float* __restrict__ d2,     // [20]
                 float*       __restrict__ agg,    // d_out [N,128] (pre-zeroed)
                 float*       __restrict__ att_agg)// [N]  (pre-zeroed)
{
    __shared__ bf16_t ldsW[HIDDEN_C * KTOT_C];  // W1^T : [n][k], 64 KB

    const int tid = threadIdx.x;

    // stage W1^T into LDS as bf16 (read coalesced over k-major W1)
    for (int idx = tid; idx < HIDDEN_C * KTOT_C; idx += 256) {
        int k = idx >> 7;        // 0..255
        int n = idx & 127;       // 0..127
        ldsW[n * KTOT_C + k] = (bf16_t)W1[idx];
    }
    __syncthreads();

    const int wave    = tid >> 5;
    const int lane    = tid & 31;
    const int halfSel = lane >> 4;      // 0: lanes 0-15, 1: lanes 16-31
    const int lidx    = lane & 15;

    const int ebase = blockIdx.x * 256 + wave * 32;   // this wave's 32 edges

    // --- per-lane edges for the two M-tiles (A row m == lidx) -------------
    int e0 = ebase + lidx;        if (e0 >= N_EDGES_C) e0 = N_EDGES_C - 1;
    int e1 = ebase + 16 + lidx;   if (e1 >= N_EDGES_C) e1 = N_EDGES_C - 1;
    const int srcE[2] = { ei[e0],             ei[e1]             };
    const int tgtE[2] = { ei[N_EDGES_C + e0], ei[N_EDGES_C + e1] };

    v8f acc[2][8] = {};              // 2 M-tiles x 8 N-tiles of 16x16 f32

    const int kA = halfSel * 16;     // A-fragment K base within 32-chunk
    const int kB = halfSel * 16;     // B-fragment K base within 32-chunk

    const float* ps0 = x + (size_t)srcE[0] * HIDDEN_C;
    const float* pt0 = x + (size_t)tgtE[0] * HIDDEN_C;
    const float* ps1 = x + (size_t)srcE[1] * HIDDEN_C;
    const float* pt1 = x + (size_t)tgtE[1] * HIDDEN_C;

    #pragma unroll 1
    for (int ks = 0; ks < 8; ++ks) {           // K = 256 in steps of 32
        const int kg  = ks * 32;
        const int off = (ks < 4) ? (ks * 32) : ((ks - 4) * 32);
        const float* r0 = ((ks < 4) ? ps0 : pt0) + off + kA;
        const float* r1 = ((ks < 4) ? ps1 : pt1) + off + kA;

        float4 a0 = *(const float4*)(r0);
        float4 a1 = *(const float4*)(r0 + 4);
        float4 a2 = *(const float4*)(r0 + 8);
        float4 a3 = *(const float4*)(r0 + 12);
        float4 b0 = *(const float4*)(r1);
        float4 b1 = *(const float4*)(r1 + 4);
        float4 b2 = *(const float4*)(r1 + 8);
        float4 b3 = *(const float4*)(r1 + 12);

        v16bf af0, af1;
        af0[0]=(bf16_t)a0.x; af0[1]=(bf16_t)a0.y; af0[2]=(bf16_t)a0.z; af0[3]=(bf16_t)a0.w;
        af0[4]=(bf16_t)a1.x; af0[5]=(bf16_t)a1.y; af0[6]=(bf16_t)a1.z; af0[7]=(bf16_t)a1.w;
        af0[8]=(bf16_t)a2.x; af0[9]=(bf16_t)a2.y; af0[10]=(bf16_t)a2.z; af0[11]=(bf16_t)a2.w;
        af0[12]=(bf16_t)a3.x; af0[13]=(bf16_t)a3.y; af0[14]=(bf16_t)a3.z; af0[15]=(bf16_t)a3.w;
        af1[0]=(bf16_t)b0.x; af1[1]=(bf16_t)b0.y; af1[2]=(bf16_t)b0.z; af1[3]=(bf16_t)b0.w;
        af1[4]=(bf16_t)b1.x; af1[5]=(bf16_t)b1.y; af1[6]=(bf16_t)b1.z; af1[7]=(bf16_t)b1.w;
        af1[8]=(bf16_t)b2.x; af1[9]=(bf16_t)b2.y; af1[10]=(bf16_t)b2.z; af1[11]=(bf16_t)b2.w;
        af1[12]=(bf16_t)b3.x; af1[13]=(bf16_t)b3.y; af1[14]=(bf16_t)b3.z; af1[15]=(bf16_t)b3.w;

        // B double-buffer: issue load of fragment t+1 before the two WMMAs
        // that consume fragment t (ds latency hidden behind WMMA issue).
        const bf16_t* brow = &ldsW[lidx * KTOT_C + kg + kB];
        v16bf bcur = *(const v16bf*)(brow);
        #pragma unroll
        for (int t = 0; t < 8; ++t) {
            v16bf bnext = bcur;
            if (t < 7) bnext = *(const v16bf*)(brow + (t + 1) * 16 * KTOT_C);
            acc[0][t] = __builtin_amdgcn_wmma_f32_16x16x32_bf16(
                false, af0, false, bcur, (short)0, acc[0][t], false, false);
            acc[1][t] = __builtin_amdgcn_wmma_f32_16x16x32_bf16(
                false, af1, false, bcur, (short)0, acc[1][t], false, false);
            bcur = bnext;
        }
    }

    // --- leaky_relu + h . W2[0:128]  (per lane: n = t*16+lidx) ------------
    float w2v[8];
    #pragma unroll
    for (int t = 0; t < 8; ++t) w2v[t] = W2[t * 16 + lidx];

    float attv[2][8];
    #pragma unroll
    for (int g = 0; g < 2; ++g) {
        float hsum[8];
        #pragma unroll
        for (int r = 0; r < 8; ++r) hsum[r] = 0.f;

        #pragma unroll
        for (int t = 0; t < 8; ++t) {
            #pragma unroll
            for (int r = 0; r < 8; ++r) {
                float h = acc[g][t][r];
                h = (h > 0.f) ? h : 0.2f * h;      // leaky_relu(0.2)
                hsum[r] += h * w2v[t];
            }
        }

        // reduce over the 16 lanes of each half (xor masks <16 stay in-half)
        #pragma unroll
        for (int r = 0; r < 8; ++r) {
            float v = hsum[r];
            v += __shfl_xor(v, 1, 32);
            v += __shfl_xor(v, 2, 32);
            v += __shfl_xor(v, 4, 32);
            v += __shfl_xor(v, 8, 32);
            hsum[r] = v;
        }

        // attention scalar per edge: att = exp(sigmoid(hsum + d2[bucket]))
        #pragma unroll
        for (int r = 0; r < 8; ++r) {
            int e  = ebase + g * 16 + r + 8 * halfSel;
            int ec = (e < N_EDGES_C) ? e : (N_EDGES_C - 1);
            int bucket = dist[ec] / 50;
            float lin  = hsum[r] + d2[bucket];
            float sg   = 1.f / (1.f + __expf(-lin));
            attv[g][r] = __expf(sg);
        }

        // att_agg scatter (one lane per edge)
        #pragma unroll
        for (int r = 0; r < 8; ++r) {
            int e = ebase + g * 16 + r + 8 * halfSel;
            if (lidx == r && e < N_EDGES_C) {
                int tg = ei[N_EDGES_C + e];
                unsafeAtomicAdd(&att_agg[tg], attv[g][r]);
            }
        }
    }

    // --- weighted feature scatter: agg[tgt] += x[src] * att ---------------
    #pragma unroll
    for (int g = 0; g < 2; ++g) {
        #pragma unroll
        for (int m = 0; m < 16; ++m) {
            float attm = __shfl(attv[g][m & 7], (m < 8) ? 0 : 16, 32);
            int srcm   = __shfl(srcE[g], m, 32);   // lane m held this edge
            int tgtm   = __shfl(tgtE[g], m, 32);
            int e = ebase + g * 16 + m;
            if (e < N_EDGES_C) {
                #pragma unroll
                for (int c = 0; c < 4; ++c) {
                    int col = c * 32 + lane;
                    float v = x[(size_t)srcm * HIDDEN_C + col] * attm;
                    unsafeAtomicAdd(&agg[(size_t)tgtm * HIDDEN_C + col], v);
                }
            }
        }
    }
}

// ---------------------------------------------------------------------------
// Kernel 3: out = relu(agg / (att_agg + 1e-6)), in place on d_out
// ---------------------------------------------------------------------------
__global__ void finalize_kernel(float* __restrict__ out,
                                const float* __restrict__ att_agg) {
    int i = blockIdx.x * blockDim.x + threadIdx.x;
    if (i < N_NODES_C * HIDDEN_C) {
        float a = att_agg[i >> 7];
        float v = out[i] / (a + 1e-6f);
        out[i] = (v > 0.f) ? v : 0.f;
    }
}

// ---------------------------------------------------------------------------
extern "C" void kernel_launch(void* const* d_in, const int* in_sizes, int n_in,
                              void* d_out, int out_size, void* d_ws, size_t ws_size,
                              hipStream_t stream) {
    const float* x     = (const float*)d_in[0];  // [N,128]
    const int*   ei    = (const int*)  d_in[1];  // [2,E]
    const int*   dist  = (const int*)  d_in[2];  // [E]
    const float* W1    = (const float*)d_in[3];  // [256,128]
    const float* W2    = (const float*)d_in[4];  // [256,1]
    const float* table = (const float*)d_in[5];  // [20,128]

    float* out     = (float*)d_out;              // [N,128], reused as agg
    float* d2      = (float*)d_ws;               // 20 floats (+pad)
    float* att_agg = d2 + 32;                    // N floats

    zero_kernel<<<512, 256, 0, stream>>>(out, N_NODES_C * HIDDEN_C,
                                         att_agg, N_NODES_C);
    d2_kernel<<<1, 32, 0, stream>>>(table, W2, d2);

    const int nBlocks = (N_EDGES_C + 255) / 256;  // 3125 blocks, 256 edges ea
    edge_kernel<<<nBlocks, 256, 0, stream>>>(x, ei, dist, W1, W2, d2,
                                             out, att_agg);

    finalize_kernel<<<(N_NODES_C * HIDDEN_C + 255) / 256, 256, 0, stream>>>(
        out, att_agg);
}
